// MemoryUnit_40389872452148
// MI455X (gfx1250) — matrix operand order repaired and checked by reading, not compiled, not run on hardware
//
#include <hip/hip_runtime.h>
#include <cstdint>
#include <cstddef>

// ---------------- types / helpers ----------------
typedef __attribute__((ext_vector_type(16))) __bf16 v16bf;
typedef __attribute__((ext_vector_type(8)))  float  v8f;

#define DEVI static __device__ __forceinline__

DEVI __bf16 f2bf(float f) {
  union { float f; unsigned u; } a; a.f = f;
  unsigned r = (a.u + 0x7FFFu + ((a.u >> 16) & 1u)) >> 16;   // RNE
  union { unsigned short s; __bf16 b; } o; o.s = (unsigned short)r;
  return o.b;
}

DEVI v8f vzero8() {
  v8f z;
#pragma unroll
  for (int i = 0; i < 8; ++i) z[i] = 0.0f;
  return z;
}

// A-fragment (16x32 bf16, row-major [M][K] source):
// lanes 0-15: M=ln, K = {kb+0..7, kb+16..23}; lanes 16-31: K = {kb+8..15, kb+24..31}
DEVI v16bf ldA(const __bf16* base, int rs, int rbase, int kbase) {
  int lane = (int)(threadIdx.x & 31u); int half = lane >> 4; int ln = lane & 15;
  const __bf16* p0 = base + (size_t)(rbase + ln) * rs + kbase + half * 8;
  const __bf16* p1 = p0 + 16;
  union { int4 q; __bf16 h[8]; } u0, u1;
  u0.q = *reinterpret_cast<const int4*>(p0);
  u1.q = *reinterpret_cast<const int4*>(p1);
  v16bf f;
#pragma unroll
  for (int i = 0; i < 8; ++i) { f[i] = u0.h[i]; f[8 + i] = u1.h[i]; }
  return f;
}

// B-fragment (32x16 bf16, source stored row-major [N][K]):
// lanes 0-15: N=ln, K = {kb+0..7, kb+16..23}; lanes 16-31: K = {kb+8..15, kb+24..31}
DEVI v16bf ldB(const __bf16* base, int rs, int nbase, int kbase) {
  int lane = (int)(threadIdx.x & 31u); int half = lane >> 4; int ln = lane & 15;
  const __bf16* p0 = base + (size_t)(nbase + ln) * rs + kbase + half * 8;
  const __bf16* p1 = p0 + 16;
  union { int4 q; __bf16 h[8]; } u0, u1;
  u0.q = *reinterpret_cast<const int4*>(p0);
  u1.q = *reinterpret_cast<const int4*>(p1);
  v16bf f;
#pragma unroll
  for (int i = 0; i < 8; ++i) { f[i] = u0.h[i]; f[8 + i] = u1.h[i]; }
  return f;
}

#define WMMA_BF16(A, Bm, Cacc) \
  __builtin_amdgcn_wmma_f32_16x16x32_bf16(false, (A), false, (Bm), (short)0, (Cacc), false, false)

// ---------------- kernel 1: pooled mean ----------------
__global__ void k_pool(const float* __restrict__ feat, float* __restrict__ pooled) {
  __shared__ float red[256];
  int bc = blockIdx.x;                       // 0..1023 = b*256+c
  const float* p = feat + (size_t)bc * 4096;
  float s = 0.f;
  for (int i = threadIdx.x; i < 4096; i += 256) s += p[i];
  red[threadIdx.x] = s; __syncthreads();
  for (int off = 128; off; off >>= 1) {
    if ((int)threadIdx.x < off) red[threadIdx.x] += red[threadIdx.x + off];
    __syncthreads();
  }
  if (threadIdx.x == 0) pooled[bc] = red[0] * (1.0f / 4096.0f);
}

// ---------------- kernel 2: MLP -> att, bias ----------------
__global__ void k_mlp(const float* __restrict__ pooled,
                      const float* __restrict__ fc1_w, const float* __restrict__ fc2_w,
                      const float* __restrict__ fc2_b, const float* __restrict__ dyn_b,
                      float* __restrict__ att_g, float* __restrict__ bias_g) {
  __shared__ float hL[260];
  __shared__ float lg[16];
  __shared__ float attL[16];
  int t = threadIdx.x;
  for (int idx = t; idx < 260; idx += 256) {
    int b = idx / 65, j = idx % 65;
    float s = 0.f;
    for (int c = 0; c < 256; ++c) s += pooled[b * 256 + c] * fc1_w[j * 256 + c];
    hL[idx] = fmaxf(s, 0.f);
  }
  __syncthreads();
  if (t < 16) {
    int b = t >> 2, k = t & 3;
    float s = fc2_b[k];
    for (int j = 0; j < 65; ++j) s += hL[b * 65 + j] * fc2_w[k * 65 + j];
    lg[t] = s / 34.0f;
  }
  __syncthreads();
  if (t < 4) {
    float m = -1e30f;
    for (int k = 0; k < 4; ++k) m = fmaxf(m, lg[t * 4 + k]);
    float sum = 0.f, e[4];
    for (int k = 0; k < 4; ++k) { e[k] = __expf(lg[t * 4 + k] - m); sum += e[k]; }
    for (int k = 0; k < 4; ++k) { float a = e[k] / sum; attL[t * 4 + k] = a; att_g[t * 4 + k] = a; }
  }
  __syncthreads();
  for (int idx = t; idx < 2048; idx += 256) {
    int b = idx >> 9, c2 = idx & 511;
    float s = 0.f;
    for (int k = 0; k < 4; ++k) s += attL[b * 4 + k] * dyn_b[k * 512 + c2];
    bias_g[idx] = s;
  }
}

// ---------------- kernel 2b: wq -> bf16 ----------------
__global__ void k_cvt_wq(const float* __restrict__ wq, __bf16* __restrict__ wq_bf) {
  int i = blockIdx.x * 256 + threadIdx.x;     // 65536
  wq_bf[i] = f2bf(wq[i]);
}

// ---------------- kernel 3: att-folded weight prep ----------------
// effW[rs][m=b*512+c2][ci] = sum_k att[b][k] * dyn_w[k][c2][ci][r][s]
__global__ void k_wprep(const float* __restrict__ dyn_w, const float* __restrict__ att_g,
                        __bf16* __restrict__ effW) {
  size_t o = (size_t)blockIdx.x * 256 + threadIdx.x;   // < 9*2048*256
  int rs = (int)(o / 524288u);
  int rem = (int)(o & 524287u);
  int m = rem >> 8, ci = rem & 255;
  int b = m >> 9, c2 = m & 511;
  float s = 0.f;
#pragma unroll
  for (int k = 0; k < 4; ++k)
    s += att_g[b * 4 + k] * dyn_w[(size_t)((k * 512 + c2) * 256 + ci) * 9 + rs];
  effW[o] = f2bf(s);
}

// ---------------- kernel 4: conv 3x3 as 9 shifted WMMA GEMMs ----------------
// key channels (c2<256)  -> keyT[b][p][d] (pixel-major, pre-transposed for attention)
// value channels         -> vB[b][dv][p]  (channel-major, direct B-fragment layout)
// grid (16 mtiles of 128 ch, 64 rows y), 256 thr (8 waves)
__global__ void k_conv(const float* __restrict__ mb, const __bf16* __restrict__ effW,
                       const float* __restrict__ bias_g,
                       __bf16* __restrict__ keyT, __bf16* __restrict__ vB) {
  __shared__ alignas(16) __bf16 mbT[66 * 32];   // [xi=x+1][ci]
  int mtile = blockIdx.x, y = blockIdx.y;
  int tid = threadIdx.x, wave = tid >> 5, lane = tid & 31;
  int half = lane >> 4, ln = lane & 15;
  v8f acc[4];
#pragma unroll
  for (int i = 0; i < 4; ++i) acc[i] = vzero8();

  for (int r = 0; r < 3; ++r) {
    int yy = y + r - 1;
    if (yy < 0 || yy >= 64) continue;          // uniform
    for (int cic = 0; cic < 8; ++cic) {
      __syncthreads();
      for (int i = tid; i < 66 * 32; i += 256) {
        int ci = i & 31, xi = i >> 5, x = xi - 1;
        float v = (x >= 0 && x < 64) ? mb[(size_t)(cic * 32 + ci) * 4096 + yy * 64 + x] : 0.f;
        mbT[xi * 32 + ci] = f2bf(v);
      }
      __syncthreads();
      int rowb = mtile * 128 + wave * 16;
#pragma unroll
      for (int s = 0; s < 3; ++s) {
        v16bf aW = ldA(effW + (size_t)(r * 3 + s) * 524288, 256, rowb, cic * 32);
        v16bf bMv[4];
#pragma unroll
        for (int sub = 0; sub < 4; ++sub) bMv[sub] = ldB(mbT, 32, sub * 16 + s, 0);
#pragma unroll
        for (int sub = 0; sub < 4; ++sub) acc[sub] = WMMA_BF16(aW, bMv[sub], acc[sub]);
      }
    }
  }
#pragma unroll
  for (int rr = 0; rr < 8; ++rr) {
    int m = mtile * 128 + wave * 16 + rr + half * 8;
    int b = m >> 9, c2 = m & 511;
    float bs = bias_g[m];
#pragma unroll
    for (int sub = 0; sub < 4; ++sub) {
      int p = y * 64 + sub * 16 + ln;
      __bf16 v = f2bf(acc[sub][rr] + bs);
      if (c2 < 256) keyT[((size_t)b * 4096 + p) * 256 + c2] = v;        // transposed key
      else          vB[((size_t)b * 256 + (c2 - 256)) * 4096 + p] = v;  // channel-major value
    }
  }
}

// ---------------- kernel 5: query projection (WMMA) ----------------
// q_bf[b][p][o] = sum_c feat[b][c][p]*wq[o][c] + bq[o]
// grid (32 ptiles of 128, 4 ntiles, 4 b), 256 thr (8 waves)
__global__ void k_qproj(const float* __restrict__ feat, const __bf16* __restrict__ wq_bf,
                        const float* __restrict__ bq, __bf16* __restrict__ q_bf) {
  __shared__ alignas(16) __bf16 FtT[128 * 32];   // [p_local][c_local]
  int pt = blockIdx.x, nt = blockIdx.y, b = blockIdx.z;
  int tid = threadIdx.x, wave = tid >> 5, lane = tid & 31;
  int half = lane >> 4, ln = lane & 15;
  v8f acc[4];
#pragma unroll
  for (int i = 0; i < 4; ++i) acc[i] = vzero8();

  for (int cc = 0; cc < 8; ++cc) {
    __syncthreads();
    for (int i = tid; i < 128 * 32; i += 256) {
      int cl = i & 31, pl = i >> 5;
      FtT[pl * 32 + cl] = f2bf(feat[((size_t)b * 256 + cc * 32 + cl) * 4096 + pt * 128 + pl]);
    }
    __syncthreads();
    v16bf aF = ldA(FtT, 32, wave * 16, 0);
    v16bf bWv[4];
#pragma unroll
    for (int sub = 0; sub < 4; ++sub) bWv[sub] = ldB(wq_bf, 256, nt * 64 + sub * 16, cc * 32);
#pragma unroll
    for (int sub = 0; sub < 4; ++sub) acc[sub] = WMMA_BF16(aF, bWv[sub], acc[sub]);
  }
#pragma unroll
  for (int rr = 0; rr < 8; ++rr) {
    int p = pt * 128 + wave * 16 + rr + half * 8;
#pragma unroll
    for (int sub = 0; sub < 4; ++sub) {
      int o = nt * 64 + sub * 16 + ln;
      q_bf[((size_t)b * 4096 + p) * 256 + o] = f2bf(acc[sub][rr] + bq[o]);
    }
  }
}

// ---------------- kernel 6: flash attention (WMMA) ----------------
// grid (32 itiles of 128 rows, 4 batch), 256 thr (8 waves, 16 query rows each)
__global__ void k_attn(const __bf16* __restrict__ q_bf, const __bf16* __restrict__ keyT,
                       const __bf16* __restrict__ vB, float* __restrict__ out) {
  __shared__ alignas(16) __bf16 Kt[32 * 256];    // [j_local][d]
  __shared__ alignas(16) __bf16 Vt[256 * 32];    // [d][j_local]
  __shared__ alignas(16) __bf16 Pst[8 * 16 * 32];
  int it = blockIdx.x, b = blockIdx.y;
  int tid = threadIdx.x, wave = tid >> 5, lane = tid & 31;
  int half = lane >> 4, ln = lane & 15;
  int i0 = it * 128 + wave * 16;

  const __bf16* qb = q_bf + (size_t)b * 4096 * 256;   // [p][d]
  const __bf16* kb = keyT + (size_t)b * 4096 * 256;   // [j][d]
  const __bf16* vb = vB + (size_t)b * 256 * 4096;     // [d][j]

  // Q fragments held in registers for the whole kernel (re-used 128x)
  v16bf aQ[8];
#pragma unroll
  for (int dc = 0; dc < 8; ++dc) aQ[dc] = ldA(qb, 256, i0, dc * 32);

  v8f O[16];
#pragma unroll
  for (int i = 0; i < 16; ++i) O[i] = vzero8();
  float mrow[8], lrow[8];
#pragma unroll
  for (int r = 0; r < 8; ++r) { mrow[r] = -1e30f; lrow[r] = 0.f; }

  for (int jt = 0; jt < 128; ++jt) {
    int j0 = jt * 32;
    __syncthreads();
    // K tile: contiguous b128 copies (pre-transposed in global)
    for (int i = tid; i < 1024; i += 256) {            // 32 rows x 32 int4
      int row = i >> 5, q4 = i & 31;
      *reinterpret_cast<int4*>(&Kt[row * 256 + q4 * 8]) =
          *reinterpret_cast<const int4*>(&kb[(size_t)(j0 + row) * 256 + q4 * 8]);
    }
    // V tile: contiguous b128 copies (channel-major)
    for (int i = tid; i < 1024; i += 256) {            // 256 rows x 4 int4
      int d = i >> 2, q4 = i & 3;
      *reinterpret_cast<int4*>(&Vt[d * 32 + q4 * 8]) =
          *reinterpret_cast<const int4*>(&vb[(size_t)d * 4096 + j0 + q4 * 8]);
    }
    // prefetch next tile into cache while we compute
    if (jt < 127) {
      int jn = j0 + 32;
      if (tid < 32) __builtin_prefetch(&kb[(size_t)(jn + tid) * 256], 0, 1);
      __builtin_prefetch(&vb[(size_t)tid * 4096 + jn], 0, 1);
    }
    __syncthreads();

    v8f S0 = vzero8(), S1 = vzero8();
#pragma unroll
    for (int dc = 0; dc < 8; ++dc) {
      v16bf bK0 = ldB(Kt, 256, 0, dc * 32);
      v16bf bK1 = ldB(Kt, 256, 16, dc * 32);
      S0 = WMMA_BF16(aQ[dc], bK0, S0);
      S1 = WMMA_BF16(aQ[dc], bK1, S1);
    }

    float p0[8], p1[8], corr[8];
#pragma unroll
    for (int r = 0; r < 8; ++r) {
      float tm = fmaxf(S0[r], S1[r]) * 0.25f;
#pragma unroll
      for (int msk = 1; msk < 16; msk <<= 1) tm = fmaxf(tm, __shfl_xor(tm, msk, 32));
      float mn = fmaxf(mrow[r], tm);
      corr[r] = __expf(mrow[r] - mn);
      mrow[r] = mn;
      p0[r] = __expf(S0[r] * 0.25f - mn);
      p1[r] = __expf(S1[r] * 0.25f - mn);
      float ts = p0[r] + p1[r];
#pragma unroll
      for (int msk = 1; msk < 16; msk <<= 1) ts += __shfl_xor(ts, msk, 32);
      lrow[r] = lrow[r] * corr[r] + ts;
    }
#pragma unroll
    for (int dt = 0; dt < 16; ++dt) {
#pragma unroll
      for (int r = 0; r < 8; ++r) O[dt][r] *= corr[r];
    }

    __bf16* myP = &Pst[wave * 512];
#pragma unroll
    for (int r = 0; r < 8; ++r) {
      int mrw = r + half * 8;
      myP[mrw * 32 + ln]      = f2bf(p0[r]);
      myP[mrw * 32 + 16 + ln] = f2bf(p1[r]);
    }
    __syncthreads();
    v16bf aP = ldA(myP, 32, 0, 0);
#pragma unroll
    for (int g = 0; g < 4; ++g) {
      v16bf bVv[4];
#pragma unroll
      for (int u = 0; u < 4; ++u) bVv[u] = ldB(Vt, 32, (g * 4 + u) * 16, 0);
#pragma unroll
      for (int u = 0; u < 4; ++u) O[g * 4 + u] = WMMA_BF16(aP, bVv[u], O[g * 4 + u]);
    }
  }

#pragma unroll
  for (int r = 0; r < 8; ++r) {
    float inv = 1.0f / lrow[r];
    int p = i0 + r + half * 8;
#pragma unroll
    for (int dt = 0; dt < 16; ++dt) {
      int c = dt * 16 + ln;
      out[((size_t)b * 256 + c) * 4096 + p] = O[dt][r] * inv;
    }
  }
}

// ---------------- host launcher ----------------
extern "C" void kernel_launch(void* const* d_in, const int* in_sizes, int n_in,
                              void* d_out, int out_size, void* d_ws, size_t ws_size,
                              hipStream_t stream) {
  const float* feature = (const float*)d_in[0];
  const float* mb      = (const float*)d_in[1];
  const float* wq      = (const float*)d_in[2];
  const float* bq      = (const float*)d_in[3];
  const float* fc1_w   = (const float*)d_in[4];
  const float* fc2_w   = (const float*)d_in[5];
  const float* fc2_b   = (const float*)d_in[6];
  const float* dyn_w   = (const float*)d_in[7];
  const float* dyn_b   = (const float*)d_in[8];
  float* out = (float*)d_out;

  char* ws = (char*)d_ws;
  float*  pooled = (float*)(ws);                       // 1024 f32
  float*  att_g  = (float*)(ws + 4096);                // 16 f32
  float*  bias_g = (float*)(ws + 4096 + 256);          // 2048 f32
  __bf16* wq_bf  = (__bf16*)(ws + 16384);              // 65536 bf16
  __bf16* effW   = (__bf16*)(ws + 147456);             // 9*2048*256 bf16 -> ends 9584640
  __bf16* keyT   = (__bf16*)(ws + 9584640);            // 4*4096*256 bf16 -> ends 17973248
  __bf16* vB     = (__bf16*)(ws + 17973248);           // 4*256*4096 bf16 -> ends 26361856
  __bf16* q_bf   = (__bf16*)(ws + 26361856);           // 4*4096*256 bf16 -> ends 34750464

  k_pool  <<<1024, 256, 0, stream>>>(feature, pooled);
  k_mlp   <<<1, 256, 0, stream>>>(pooled, fc1_w, fc2_w, fc2_b, dyn_b, att_g, bias_g);
  k_cvt_wq<<<256, 256, 0, stream>>>(wq, wq_bf);
  k_wprep <<<18432, 256, 0, stream>>>(dyn_w, att_g, effW);
  k_conv  <<<dim3(16, 64), 256, 0, stream>>>(mb, effW, bias_g, keyT, vB);
  k_qproj <<<dim3(32, 4, 4), 256, 0, stream>>>(feature, wq_bf, bq, q_bf);
  k_attn  <<<dim3(32, 4), 256, 0, stream>>>(q_bf, keyT, vB, out);
}
